// Str2Str_18399639896108
// MI455X (gfx1250) — matrix operand order repaired and checked by reading, not compiled, not run on hardware
//
#include <hip/hip_runtime.h>

typedef __attribute__((ext_vector_type(16))) __bf16 v16bf;
typedef __attribute__((ext_vector_type(8)))  float  v8f;

#define EPSF 1e-5f

// native fptrunc -> v_cvt_pk_bf16_f32 on gfx1250 (RNE)
__device__ __forceinline__ __bf16 f2bf(float f) { return (__bf16)f; }

__device__ __forceinline__ v8f zero8() {
    v8f z;
#pragma unroll
    for (int i = 0; i < 8; ++i) z[i] = 0.0f;
    return z;
}

__device__ __forceinline__ v16bf ldw(const __bf16* p) {
    v16bf r; __builtin_memcpy(&r, p, sizeof(r)); return r;   // 2x ds_load_b128
}

// ---------------------------------------------------------------------------
// Kernel 0: msa layernorm + seq-0 attention + node = LN([msa_w,seq1hot]@Wx+bx)
// one block (64 threads) per (b,l)
// ---------------------------------------------------------------------------
__global__ __launch_bounds__(64)
void k_node(const float* __restrict__ msa, const float* __restrict__ seq1hot,
            const float* __restrict__ g_msa, const float* __restrict__ b_msa,
            const float* __restrict__ Wq, const float* __restrict__ bq,
            const float* __restrict__ Wk, const float* __restrict__ bk,
            const float* __restrict__ Wx, const float* __restrict__ bx,
            const float* __restrict__ g_node, const float* __restrict__ b_node,
            float* __restrict__ node)
{
    const int Nn = 32, L = 256, D = 64;
    const int b = blockIdx.x >> 8;
    const int l = blockIdx.x & 255;
    const int t = threadIdx.x;

    __shared__ float mn[32][64];
    __shared__ float kkv[32][64];
    __shared__ float red[64], red2[64];
    __shared__ float qv[64];
    __shared__ float attn[32];
    __shared__ float x85[96];

    // layernorm each msa row (over d=64)
    for (int n = 0; n < Nn; ++n) {
        float v = msa[(((size_t)b * Nn + n) * L + l) * D + t];
        red[t] = v; red2[t] = v * v;
        __syncthreads();
        for (int s = 32; s > 0; s >>= 1) {
            if (t < s) { red[t] += red[t + s]; red2[t] += red2[t + s]; }
            __syncthreads();
        }
        float mu = red[0] * (1.0f / 64.0f);
        float var = red2[0] * (1.0f / 64.0f) - mu * mu;
        float rs = rsqrtf(var + EPSF);
        mn[n][t] = (v - mu) * rs * g_msa[t] + b_msa[t];
        __syncthreads();
    }

    // q from sequence 0, keys for all n
    float q = bq[t];
    for (int k = 0; k < 64; ++k) q += mn[0][k] * Wq[k * 64 + t];
    qv[t] = q * 0.125f;  // / sqrt(64)
    for (int n = 0; n < Nn; ++n) {
        float s = bk[t];
        for (int k = 0; k < 64; ++k) s += mn[n][k] * Wk[k * 64 + t];
        kkv[n][t] = s;
    }
    __syncthreads();

    if (t < 32) {
        float s = 0.0f;
        for (int d = 0; d < 64; ++d) s += qv[d] * kkv[t][d];
        attn[t] = s;
    }
    __syncthreads();
    if (t == 0) {
        float mx = attn[0];
        for (int n = 1; n < 32; ++n) mx = fmaxf(mx, attn[n]);
        float sm = 0.0f;
        for (int n = 0; n < 32; ++n) { float e = __expf(attn[n] - mx); attn[n] = e; sm += e; }
        float inv = 1.0f / sm;
        for (int n = 0; n < 32; ++n) attn[n] *= inv;
    }
    __syncthreads();

    // msa_w then concat seq1hot
    float w = 0.0f;
    for (int n = 0; n < 32; ++n) w += attn[n] * mn[n][t];
    x85[t] = w;
    if (t < 21) x85[64 + t] = seq1hot[((size_t)b * L + l) * 21 + t];
    __syncthreads();

    float p = 0.0f;
    if (t < 32) {
        p = bx[t];
        for (int k = 0; k < 85; ++k) p += x85[k] * Wx[k * 32 + t];
        red[t] = p; red2[t] = p * p;
    }
    __syncthreads();
    for (int s = 16; s > 0; s >>= 1) {
        if (t < s) { red[t] += red[t + s]; red2[t] += red2[t + s]; }
        __syncthreads();
    }
    float mu = red[0] * (1.0f / 32.0f);
    float var = red2[0] * (1.0f / 32.0f) - mu * mu;
    float rs = rsqrtf(var + EPSF);
    if (t < 32)
        node[((size_t)b * L + l) * 32 + t] = (p - mu) * rs * g_node[t] + b_node[t];
}

// ---------------------------------------------------------------------------
// Kernel 1: geometry: D, rank-based top-k mask (stable), sep<KMIN, dhat, deg
// One block (256 threads) per (b,i); thread j handles column j.
// ---------------------------------------------------------------------------
__global__ __launch_bounds__(256)
void k_geom(const float* __restrict__ xyz, const int* __restrict__ idx,
            const int* __restrict__ tkp,
            float* __restrict__ maskA, float* __restrict__ distA,
            float* __restrict__ dhatA, float* __restrict__ degA)
{
    const int L = 256;
    const int b = blockIdx.x >> 8;
    const int i = blockIdx.x & 255;
    const int j = threadIdx.x;
    __shared__ float Dsh[256];
    __shared__ int msum[256];

    float cix = xyz[((size_t)(b * L + i) * 3 + 1) * 3 + 0];
    float ciy = xyz[((size_t)(b * L + i) * 3 + 1) * 3 + 1];
    float ciz = xyz[((size_t)(b * L + i) * 3 + 1) * 3 + 2];
    float cjx = xyz[((size_t)(b * L + j) * 3 + 1) * 3 + 0];
    float cjy = xyz[((size_t)(b * L + j) * 3 + 1) * 3 + 1];
    float cjz = xyz[((size_t)(b * L + j) * 3 + 1) * 3 + 2];
    float dx = cjx - cix, dy = cjy - ciy, dz = cjz - ciz;   // diff[b,i,j] = ca[j]-ca[i]
    float dd = sqrtf(dx * dx + dy * dy + dz * dz);
    float Dv = dd + ((i == j) ? 999.9f : 0.0f);
    Dsh[j] = Dv;
    __syncthreads();

    int rank = 0;
    for (int jj = 0; jj < L; ++jj) {
        float o = Dsh[jj];
        rank += (o < Dv) || (o == Dv && jj < j);
    }
    int k = tkp[0]; if (k > L) k = L;
    float sep = fabsf((float)(idx[b * L + j] - idx[b * L + i])) + ((i == j) ? 999.9f : 0.0f);
    int m = ((rank < k) || (sep < 9.0f)) ? 1 : 0;   // KMIN = 9

    size_t base = ((size_t)(b * L + i)) * L + j;
    maskA[base] = (float)m;
    distA[base] = dd;
    float inv = 1.0f / (dd + 1e-8f);
    dhatA[base * 3 + 0] = dx * inv;
    dhatA[base * 3 + 1] = dy * inv;
    dhatA[base * 3 + 2] = dz * inv;

    msum[j] = m;
    __syncthreads();
    for (int s = 128; s > 0; s >>= 1) {
        if (j < s) msum[j] += msum[j + s];
        __syncthreads();
    }
    if (j == 0) degA[b * L + i] = fmaxf((float)msum[0], 1.0f);
}

// ---------------------------------------------------------------------------
// Kernel 2: fused WMMA chain per (b,i):
//   pairLN -> edge -> a -> {m0, c1|c2} -> state/offset
// block = 128 threads (4 waves); each wave owns 4 tiles of 16 j's.
// pair is read exactly ONCE from HBM (the dominant traffic).
// All 16 weight WMMA-B operands are staged in LDS in per-lane register layout
// (computed by wave 0 once) to keep VGPR count < 256.
// ---------------------------------------------------------------------------
// Weight operand slots in wlds: We(c,nt)=c*2+nt [0..7], Wa(c,nt)=8+c*2+nt [8..11],
// W0(c)=12+c [12..14], Wc=15.
__global__ __launch_bounds__(128)
void k_fused(const float* __restrict__ pair, const float* __restrict__ node,
             const float* __restrict__ maskA, const float* __restrict__ distA,
             const float* __restrict__ dhatA, const float* __restrict__ degA,
             const float* __restrict__ xyz,
             const float* __restrict__ g_pair, const float* __restrict__ b_pair,
             const float* __restrict__ We, const float* __restrict__ be,
             const float* __restrict__ g_edge, const float* __restrict__ b_edge,
             const float* __restrict__ Wa, const float* __restrict__ ba,
             const float* __restrict__ W0, const float* __restrict__ b0,
             const float* __restrict__ Wc1, const float* __restrict__ Wc2,
             float* __restrict__ out_xyz, float* __restrict__ out_state)
{
    const int L = 256;
    const int b = blockIdx.x >> 8;
    const int i = blockIdx.x & 255;
    const int tid = threadIdx.x;
    const int lane = tid & 31;
    const int wave = tid >> 5;
    const int n16 = lane & 15;          // N column (B/C/D) or M row (A)
    const int kh = (lane >> 4) & 1;     // lane half selector

    __shared__ __align__(32) __bf16 wlds[16][32][16];   // 16 KB weight operands
    __shared__ float lds_edge[4][16][33];
    __shared__ float lds_a[4][16][33];
    __shared__ float lds_c[4][16][12];
    __shared__ float lds_dist[4][16];
    __shared__ float lds_mask[4][16];
    __shared__ float g_lds[128], bb_lds[128];
    __shared__ float s_state[16];
    __shared__ float s_off[9];

    if (tid < 16) s_state[tid] = 0.0f;
    if (tid < 9)  s_off[tid] = 0.0f;
    g_lds[tid]  = g_pair[tid];      // blockDim == 128 == d_pair
    bb_lds[tid] = b_pair[tid];

    // ---- wave 0 stages all weight B-operands in per-lane register layout:
    //      lane -> N = n16, K = chunk*32 + ((e&8)<<1) + kh*8 + (e&7)
    if (wave == 0) {
#pragma unroll
        for (int c = 0; c < 4; ++c)
#pragma unroll
            for (int nt = 0; nt < 2; ++nt)
#pragma unroll
                for (int e = 0; e < 16; ++e) {
                    int K = c * 32 + ((e & 8) << 1) + kh * 8 + (e & 7);
                    wlds[c * 2 + nt][lane][e] = f2bf(We[K * 32 + nt * 16 + n16]);
                }
#pragma unroll
        for (int c = 0; c < 2; ++c)
#pragma unroll
            for (int nt = 0; nt < 2; ++nt)
#pragma unroll
                for (int e = 0; e < 16; ++e) {
                    int K = c * 32 + ((e & 8) << 1) + kh * 8 + (e & 7);
                    wlds[8 + c * 2 + nt][lane][e] =
                        (K < 33) ? f2bf(Wa[K * 32 + nt * 16 + n16]) : f2bf(0.0f);
                }
#pragma unroll
        for (int c = 0; c < 3; ++c)
#pragma unroll
            for (int e = 0; e < 16; ++e) {
                int K = c * 32 + ((e & 8) << 1) + kh * 8 + (e & 7);
                wlds[12 + c][lane][e] = f2bf(W0[K * 16 + n16]);
            }
#pragma unroll
        for (int e = 0; e < 16; ++e) {
            int K = ((e & 8) << 1) + kh * 8 + (e & 7);
            float v = 0.0f;
            if (n16 < 3)       v = Wc1[K * 3 + n16];
            else if (n16 < 12) v = Wc2[K * 9 + (n16 - 3)];
            wlds[15][lane][e] = f2bf(v);
        }
    }

    const float beL0 = be[n16],        beL1 = be[16 + n16];
    const float geL0 = g_edge[n16],    geL1 = g_edge[16 + n16];
    const float bEL0 = b_edge[n16],    bEL1 = b_edge[16 + n16];
    const float baL0 = ba[n16],        baL1 = ba[16 + n16];
    const float b0L  = b0[n16];

    float offp[9];
#pragma unroll
    for (int q = 0; q < 9; ++q) offp[q] = 0.0f;

    __syncthreads();

    for (int t = 0; t < 4; ++t) {
        const int tt = t * 4 + wave;
        const int j0 = tt * 16;
        const int rowm = n16;   // A-matrix row for this lane

        // ---- fused pair layernorm + A(pair_n) build.
        // Row rowm's 128 features split across lane pair (rowm,kh=0)/(rowm,kh=1).
        const float* prow = pair + (((size_t)(b * L + i)) * L + (j0 + rowm)) * 128;

        // pass 1: per-lane partial sum / sumsq over this lane's 64 elements
        float s = 0.0f, s2 = 0.0f;
#pragma unroll
        for (int c = 0; c < 4; ++c)
#pragma unroll
            for (int e = 0; e < 16; ++e) {
                int K = c * 32 + ((e & 8) << 1) + kh * 8 + (e & 7);
                float v = prow[K];
                s += v; s2 += v * v;
            }
        s  += __shfl_xor(s,  16, 32);
        s2 += __shfl_xor(s2, 16, 32);
        float mu = s * (1.0f / 128.0f);
        float var = s2 * (1.0f / 128.0f) - mu * mu;
        float prs = rsqrtf(var + EPSF);

        // pass 2: reload (cache-hot), normalize, convert to bf16 A operand
        v16bf Ap[4];
#pragma unroll
        for (int c = 0; c < 4; ++c) {
            v16bf av;
#pragma unroll
            for (int e = 0; e < 16; ++e) {
                int K = c * 32 + ((e & 8) << 1) + kh * 8 + (e & 7);
                float v = prow[K];
                av[e] = f2bf((v - mu) * prs * g_lds[K] + bb_lds[K]);
            }
            Ap[c] = av;
        }

        // ---- edge GEMM: (16x128)@(128x32)
        v8f C0 = zero8(), C1 = zero8();
#pragma unroll
        for (int c = 0; c < 4; ++c) {
            C0 = __builtin_amdgcn_wmma_f32_16x16x32_bf16(false, Ap[c], false,
                     ldw(&wlds[c * 2 + 0][lane][0]), (short)0, C0, false, false);
            C1 = __builtin_amdgcn_wmma_f32_16x16x32_bf16(false, Ap[c], false,
                     ldw(&wlds[c * 2 + 1][lane][0]), (short)0, C1, false, false);
        }

        // ---- bias + layernorm over 32 features (row = v + kh*8), 16-lane shfl reduce
#pragma unroll
        for (int v = 0; v < 8; ++v) {
            float x0 = C0[v] + beL0;
            float x1 = C1[v] + beL1;
            float sr = x0 + x1, sr2 = x0 * x0 + x1 * x1;
#pragma unroll
            for (int mlt = 1; mlt < 16; mlt <<= 1) {
                sr  += __shfl_xor(sr,  mlt, 16);
                sr2 += __shfl_xor(sr2, mlt, 16);
            }
            float emu = sr * (1.0f / 32.0f);
            float evar = sr2 * (1.0f / 32.0f) - emu * emu;
            float ers = rsqrtf(evar + EPSF);
            int mrow = v + kh * 8;
            lds_edge[wave][mrow][n16]      = (x0 - emu) * ers * geL0 + bEL0;
            lds_edge[wave][mrow][16 + n16] = (x1 - emu) * ers * geL1 + bEL1;
        }
        if (lane < 16) {
            size_t base = ((size_t)(b * L + i)) * L + (j0 + lane);
            lds_dist[wave][lane] = distA[base];
            lds_mask[wave][lane] = maskA[base];
        }
        __syncthreads();

        // ---- A([edge,dist] padded to K=64) via LDS transpose
        v16bf Aa[2];
#pragma unroll
        for (int c = 0; c < 2; ++c) {
            v16bf av;
#pragma unroll
            for (int e = 0; e < 16; ++e) {
                int K = c * 32 + ((e & 8) << 1) + kh * 8 + (e & 7);
                float val;
                if (K < 32)       val = lds_edge[wave][rowm][K];
                else if (K == 32) val = lds_dist[wave][rowm];
                else              val = 0.0f;
                av[e] = f2bf(val);
            }
            Aa[c] = av;
        }
        v8f Ca0 = zero8(), Ca1 = zero8();
#pragma unroll
        for (int c = 0; c < 2; ++c) {
            Ca0 = __builtin_amdgcn_wmma_f32_16x16x32_bf16(false, Aa[c], false,
                      ldw(&wlds[8 + c * 2 + 0][lane][0]), (short)0, Ca0, false, false);
            Ca1 = __builtin_amdgcn_wmma_f32_16x16x32_bf16(false, Aa[c], false,
                      ldw(&wlds[8 + c * 2 + 1][lane][0]), (short)0, Ca1, false, false);
        }
#pragma unroll
        for (int v = 0; v < 8; ++v) {
            int mrow = v + kh * 8;
            lds_a[wave][mrow][n16]      = fmaxf(Ca0[v] + baL0, 0.0f);
            lds_a[wave][mrow][16 + n16] = fmaxf(Ca1[v] + baL1, 0.0f);
        }
        __syncthreads();

        // ---- A([a, h_i, h_j]) : chunk0 from LDS, chunk1/2 from node (f32->bf16)
        v16bf Am[3];
        {
            v16bf a0, a1, a2;
#pragma unroll
            for (int e = 0; e < 16; ++e) {
                int kk = ((e & 8) << 1) + kh * 8 + (e & 7);   // 0..31
                a0[e] = f2bf(lds_a[wave][rowm][kk]);
                a1[e] = f2bf(node[((size_t)(b * L + i)) * 32 + kk]);
                a2[e] = f2bf(node[((size_t)(b * L + (j0 + rowm))) * 32 + kk]);
            }
            Am[0] = a0; Am[1] = a1; Am[2] = a2;
        }
        v8f Cm = zero8();
#pragma unroll
        for (int c = 0; c < 3; ++c)
            Cm = __builtin_amdgcn_wmma_f32_16x16x32_bf16(false, Am[c], false,
                     ldw(&wlds[12 + c][lane][0]), (short)0, Cm, false, false);
        float m0v[8];
#pragma unroll
        for (int v = 0; v < 8; ++v) m0v[v] = fmaxf(Cm[v] + b0L, 0.0f);

        // ---- c = a @ [Wc1|Wc2] (N padded 12->16)
        v8f Cc = __builtin_amdgcn_wmma_f32_16x16x32_bf16(false, Am[0], false,
                     ldw(&wlds[15][lane][0]), (short)0, zero8(), false, false);

        // ---- state partial: lane has feature n16, rows v+kh*8
        float sp = 0.0f;
#pragma unroll
        for (int v = 0; v < 8; ++v) sp += lds_mask[wave][v + kh * 8] * m0v[v];
        sp += __shfl_xor(sp, 16, 32);
        if (lane < 16) atomicAdd(&s_state[n16], sp);

        // ---- transpose c tile to LDS, then lanes<16 build v_msg and accumulate
        if (n16 < 12) {
#pragma unroll
            for (int v = 0; v < 8; ++v) lds_c[wave][v + kh * 8][n16] = Cc[v];
        }
        __syncthreads();

        if (lane < 16) {
            const int m = lane;
            const int j = j0 + m;
            float mval = lds_mask[wave][m];
            size_t dbase = (((size_t)(b * L + i)) * L + j) * 3;
            float dh[3] = { dhatA[dbase + 0], dhatA[dbase + 1], dhatA[dbase + 2] };
            float cax = xyz[((size_t)(b * L + j) * 3 + 1) * 3 + 0];
            float cay = xyz[((size_t)(b * L + j) * 3 + 1) * 3 + 1];
            float caz = xyz[((size_t)(b * L + j) * 3 + 1) * 3 + 2];
            float l1[3][3];
#pragma unroll
            for (int cc = 0; cc < 3; ++cc) {
                l1[cc][0] = xyz[((size_t)(b * L + j) * 3 + cc) * 3 + 0] - cax;
                l1[cc][1] = xyz[((size_t)(b * L + j) * 3 + cc) * 3 + 1] - cay;
                l1[cc][2] = xyz[((size_t)(b * L + j) * 3 + cc) * 3 + 2] - caz;
            }
#pragma unroll
            for (int o = 0; o < 3; ++o) {
                float c1o = lds_c[wave][m][o];
                float c20 = lds_c[wave][m][3 + o * 3 + 0];
                float c21 = lds_c[wave][m][3 + o * 3 + 1];
                float c22 = lds_c[wave][m][3 + o * 3 + 2];
#pragma unroll
                for (int x = 0; x < 3; ++x) {
                    float vm = c1o * dh[x] + c20 * l1[0][x] + c21 * l1[1][x] + c22 * l1[2][x];
                    offp[o * 3 + x] += mval * vm;
                }
            }
        }
        __syncthreads();
    }

    if (lane < 16) {
#pragma unroll
        for (int q = 0; q < 9; ++q) atomicAdd(&s_off[q], offp[q]);
    }
    __syncthreads();

    float dg = degA[b * L + i];
    float invdg = 1.0f / dg;
    if (tid < 16)
        out_state[((size_t)(b * L + i)) * 16 + tid] = s_state[tid] * invdg;
    if (tid < 9) {
        int o = tid / 3, x = tid % 3;
        float cax = xyz[((size_t)(b * L + i) * 3 + 1) * 3 + x];
        float offCA = s_off[3 + x] * invdg;
        float val = cax + offCA + ((o == 1) ? 0.0f : s_off[o * 3 + x] * invdg);
        out_xyz[((size_t)(b * L + i) * 3 + o) * 3 + x] = val;
    }
}

// ---------------------------------------------------------------------------
extern "C" void kernel_launch(void* const* d_in, const int* in_sizes, int n_in,
                              void* d_out, int out_size, void* d_ws, size_t ws_size,
                              hipStream_t stream)
{
    (void)in_sizes; (void)n_in; (void)out_size; (void)ws_size;
    const int B = 2, L = 256;

    const float* msa     = (const float*)d_in[0];
    const float* pair    = (const float*)d_in[1];
    const float* xyz     = (const float*)d_in[2];
    const float* seq1hot = (const float*)d_in[3];
    const float* g_msa   = (const float*)d_in[4];
    const float* b_msa   = (const float*)d_in[5];
    const float* g_pair  = (const float*)d_in[6];
    const float* b_pair  = (const float*)d_in[7];
    const float* Wq      = (const float*)d_in[8];
    const float* bq      = (const float*)d_in[9];
    const float* Wk      = (const float*)d_in[10];
    const float* bk      = (const float*)d_in[11];
    const float* Wx      = (const float*)d_in[12];
    const float* bx      = (const float*)d_in[13];
    const float* g_node  = (const float*)d_in[14];
    const float* b_node  = (const float*)d_in[15];
    const float* We      = (const float*)d_in[16];
    const float* be      = (const float*)d_in[17];
    const float* g_edge  = (const float*)d_in[18];
    const float* b_edge  = (const float*)d_in[19];
    const float* Wa      = (const float*)d_in[20];
    const float* ba      = (const float*)d_in[21];
    const float* W0      = (const float*)d_in[22];
    const float* b0      = (const float*)d_in[23];
    const float* Wc1     = (const float*)d_in[24];
    const float* Wc2     = (const float*)d_in[25];
    const int*   idx     = (const int*)d_in[26];
    const int*   topk    = (const int*)d_in[27];

    char* ws = (char*)d_ws;
    size_t off = 0;
    float* node  = (float*)(ws + off);  off += (size_t)B * L * 32 * sizeof(float);
    float* maskA = (float*)(ws + off);  off += (size_t)B * L * L * sizeof(float);
    float* distA = (float*)(ws + off);  off += (size_t)B * L * L * sizeof(float);
    float* dhatA = (float*)(ws + off);  off += (size_t)B * L * L * 3 * sizeof(float);
    float* degA  = (float*)(ws + off);  off += (size_t)B * L * sizeof(float);

    float* out_xyz   = (float*)d_out;                 // (B,L,3,3)
    float* out_state = out_xyz + (size_t)B * L * 9;   // (B,L,16)

    k_node<<<B * L, 64, 0, stream>>>(msa, seq1hot, g_msa, b_msa, Wq, bq, Wk, bk,
                                     Wx, bx, g_node, b_node, node);
    k_geom<<<B * L, 256, 0, stream>>>(xyz, idx, topk, maskA, distA, dhatA, degA);
    k_fused<<<B * L, 128, 0, stream>>>(pair, node, maskA, distA, dhatA, degA, xyz,
                                       g_pair, b_pair,
                                       We, be, g_edge, b_edge, Wa, ba, W0, b0,
                                       Wc1, Wc2, out_xyz, out_state);
}
